// MultiHeadAttention_55052890800538
// MI455X (gfx1250) — matrix operand order repaired and checked by reading, not compiled, not run on hardware
//
#include <hip/hip_runtime.h>

// Problem constants (match reference)
#define B_SZ     2
#define S_LEN    2048
#define D_MODEL  1024
#define N_HEADS  16
#define HEAD_DIM 64
#define FA_NW    4                // waves per flash-attn block
#define MTOT     (B_SZ * S_LEN)   // 4096 rows

// Async LDS copy engine (gfx1250): guarded so compile never regresses.
#if defined(__has_builtin)
#  if __has_builtin(__builtin_amdgcn_global_load_async_to_lds_b128)
#    define HAVE_ASYNC_LDS 1
#  endif
#endif
#ifndef HAVE_ASYNC_LDS
#  define HAVE_ASYNC_LDS 0
#endif

typedef _Float16 half8  __attribute__((ext_vector_type(8)));
typedef _Float16 half16 __attribute__((ext_vector_type(16)));
typedef float    float8 __attribute__((ext_vector_type(8)));
typedef int      int4v  __attribute__((ext_vector_type(4)));

#define AS1 __attribute__((address_space(1)))
#define AS3 __attribute__((address_space(3)))

// ---------------------------------------------------------------------------
// WMMA fragment helpers (per CDNA5 ISA 7.12.2 VGPR layouts, wave32)
//
// A-matrix 16x32 f16: lane L holds row M=L&15; half=L>>4 selects
//   K runs [8*half, 8*half+8) and [16+8*half, 16+8*half+8)  -> two b128 loads
// B-matrix 32x16 f16 (loaded from row-major B^T, i.e. [N][K]): lane L holds
//   column N=L&15; K runs [8*half, 8*half+8) and [16+8*half, 16+8*half+8) -> two b128 loads
// C/D 16x16 f32: lane L holds col N=L&15; VGPR v holds row M = 8*(L>>4)+v
// ---------------------------------------------------------------------------
__device__ __forceinline__ half16 load_frag_a(const _Float16* rowptr, int k0, int half_) {
  union { half16 v; half8 q[2]; } u;
  const _Float16* p = rowptr + k0 + 8 * half_;
  u.q[0] = *(const half8*)(p);
  u.q[1] = *(const half8*)(p + 16);
  return u.v;
}
__device__ __forceinline__ half16 load_frag_b(const _Float16* rowptr, int k0, int half_) {
  union { half16 v; half8 q[2]; } u;
  const _Float16* p = rowptr + k0 + 8 * half_;
  u.q[0] = *(const half8*)(p);
  u.q[1] = *(const half8*)(p + 16);
  return u.v;
}
__device__ __forceinline__ float8 wmma_f16(half16 a, half16 b, float8 c) {
  return __builtin_amdgcn_wmma_f32_16x16x32_f16(false, a, false, b, (short)0, c, false, false);
}

__device__ __forceinline__ void wait_async0() {
#if HAVE_ASYNC_LDS
#  if __has_builtin(__builtin_amdgcn_s_wait_asynccnt)
  __builtin_amdgcn_s_wait_asynccnt(0);
#  else
  asm volatile("s_wait_asynccnt 0" ::: "memory");
#  endif
#endif
}

// Async-copy a 32-row x 64-half K block (rows j0..j0+31 of this head) into LDS.
// 32 lanes x 8 b128 chunks = 4 KB; tracked on ASYNCcnt.
__device__ __forceinline__ void async_copy_k(const _Float16* gbase, int j0,
                                             _Float16* lbuf, int lane) {
#if HAVE_ASYNC_LDS
  const int cg = lane >> 3;         // row sub-group 0..3
  const int co = (lane & 7) * 8;    // half offset within row (16B granules)
#pragma unroll
  for (int c = 0; c < 8; c++) {
    const int r = c * 4 + cg;       // 0..31
    const _Float16* g = gbase + (size_t)(j0 + r) * D_MODEL + co;
    _Float16* l = lbuf + r * 64 + co;
    __builtin_amdgcn_global_load_async_to_lds_b128(
        (AS1 int4v*)g, (AS3 int4v*)l, 0, 0);
  }
#endif
}

// ---------------------------------------------------------------------------
// 1) f32 -> f16 conversion
// ---------------------------------------------------------------------------
__global__ void cvt_f32_f16(const float* __restrict__ in, _Float16* __restrict__ out, int n) {
  int i = blockIdx.x * blockDim.x + threadIdx.x;
  if (i < n) out[i] = (_Float16)in[i];
}

// ---------------------------------------------------------------------------
// 2) GEMM: C[M][N] (f32) = A[M][K] (f16, row-major) @ W[N][K]^T (f16, row-major)
//    256 threads = 8 waves arranged 2(M) x 4(N); wave tile 32x64:
//    8 WMMAs per k-step off 6 fragment loads, 8 independent acc chains.
//    Weights stay L2-resident (2 MB << 192 MB), so direct global loads.
// ---------------------------------------------------------------------------
__global__ void gemm_wmma(const _Float16* __restrict__ A, const _Float16* __restrict__ W,
                          float* __restrict__ C, int M, int N, int K) {
  const int wave  = threadIdx.x >> 5;
  const int lane  = threadIdx.x & 31;
  const int r16   = lane & 15;
  const int half_ = lane >> 4;
  const int m0 = (blockIdx.y * 2 + (wave >> 2)) * 32;   // wave M tile: 32 rows
  const int n0 = (blockIdx.x * 4 + (wave & 3)) * 64;    // wave N tile: 64 cols

  const _Float16* arow0 = A + (size_t)(m0 + r16) * K;
  const _Float16* arow1 = arow0 + (size_t)16 * K;
  const _Float16* brow0 = W + (size_t)(n0 + r16) * K;
  const _Float16* brow1 = brow0 + (size_t)16 * K;
  const _Float16* brow2 = brow0 + (size_t)32 * K;
  const _Float16* brow3 = brow0 + (size_t)48 * K;

  float8 acc[2][4];
#pragma unroll
  for (int i = 0; i < 2; i++)
#pragma unroll
    for (int j = 0; j < 4; j++) acc[i][j] = (float8){};

#pragma unroll 2
  for (int k0 = 0; k0 < K; k0 += 32) {
    // prefetch one k-block ahead (global_prefetch_b8; speculative, OOB-safe)
    __builtin_prefetch(arow0 + k0 + 64, 0, 1);
    __builtin_prefetch(brow0 + k0 + 64, 0, 1);

    const half16 a0 = load_frag_a(arow0, k0, half_);
    const half16 a1 = load_frag_a(arow1, k0, half_);
    const half16 b0 = load_frag_b(brow0, k0, half_);
    const half16 b1 = load_frag_b(brow1, k0, half_);
    const half16 b2 = load_frag_b(brow2, k0, half_);
    const half16 b3 = load_frag_b(brow3, k0, half_);

    acc[0][0] = wmma_f16(a0, b0, acc[0][0]);
    acc[0][1] = wmma_f16(a0, b1, acc[0][1]);
    acc[0][2] = wmma_f16(a0, b2, acc[0][2]);
    acc[0][3] = wmma_f16(a0, b3, acc[0][3]);
    acc[1][0] = wmma_f16(a1, b0, acc[1][0]);
    acc[1][1] = wmma_f16(a1, b1, acc[1][1]);
    acc[1][2] = wmma_f16(a1, b2, acc[1][2]);
    acc[1][3] = wmma_f16(a1, b3, acc[1][3]);
  }

#pragma unroll
  for (int i = 0; i < 2; i++) {
    float* crow = C + (size_t)(m0 + i * 16 + 8 * half_) * N + n0 + r16;
#pragma unroll
    for (int j = 0; j < 4; j++) {
#pragma unroll
      for (int v = 0; v < 8; v++) crow[(size_t)v * N + j * 16] = acc[i][j][v];
    }
  }
}

// ---------------------------------------------------------------------------
// 3) RoPE (reference variant) on Q,K (f32 in) -> f16 out
//    out[i]    = x[2i]*cos(t*invf[(2i)%32])   - x[2i+1]*sin(t*invf[(2i)%32])
//    out[32+i] = x[2i]*sin(t*invf[(2i+1)%32]) + x[2i+1]*cos(t*invf[(2i+1)%32])
//    invf[j] = 10000^(-j/32)
// ---------------------------------------------------------------------------
__global__ void rope_kernel(const float* __restrict__ Qf, const float* __restrict__ Kf,
                            _Float16* __restrict__ Qh, _Float16* __restrict__ Kh) {
  int idx = blockIdx.x * blockDim.x + threadIdx.x;   // over B*S*H*32
  if (idx >= B_SZ * S_LEN * N_HEADS * 32) return;
  int i = idx & 31;
  int h = (idx >> 5) & (N_HEADS - 1);
  int s = (idx >> 9) & (S_LEN - 1);
  int b = idx >> 20;
  size_t base = (size_t)(b * S_LEN + s) * D_MODEL + h * HEAD_DIM;

  float fs = (float)s;
  float te = fs * __powf(10000.0f, -(float)((2 * i) & 31) * (1.0f / 32.0f));
  float to = fs * __powf(10000.0f, -(float)((2 * i + 1) & 31) * (1.0f / 32.0f));
  float ce = __cosf(te), se = __sinf(te);
  float co = __cosf(to), so = __sinf(to);

  {
    float x1 = Qf[base + 2 * i], x2 = Qf[base + 2 * i + 1];
    Qh[base + i]      = (_Float16)(x1 * ce - x2 * se);
    Qh[base + 32 + i] = (_Float16)(x1 * so + x2 * co);
  }
  {
    float x1 = Kf[base + 2 * i], x2 = Kf[base + 2 * i + 1];
    Kh[base + i]      = (_Float16)(x1 * ce - x2 * se);
    Kh[base + 32 + i] = (_Float16)(x1 * so + x2 * co);
  }
}

// ---------------------------------------------------------------------------
// 4) V f32 [b][s][h*64+d]  ->  Vt f16 [b*H+h][d][s]   (so P@V B-frags are contiguous)
// ---------------------------------------------------------------------------
__global__ void vtrans_kernel(const float* __restrict__ Vf, _Float16* __restrict__ Vt) {
  int idx = blockIdx.x * blockDim.x + threadIdx.x;   // over B*S*D
  if (idx >= B_SZ * S_LEN * D_MODEL) return;
  int n = idx & (D_MODEL - 1);
  int s = (idx >> 10) & (S_LEN - 1);
  int b = idx >> 21;
  int h = n >> 6, d = n & 63;
  Vt[((size_t)(b * N_HEADS + h) * HEAD_DIM + d) * S_LEN + s] = (_Float16)Vf[idx];
}

// ---------------------------------------------------------------------------
// 5) Flash attention (causal).  grid=(S/16/FA_NW, B*H), block=32*FA_NW.
//    Each wave: one 16-row Q tile; stream 32-col K/V blocks to the diagonal.
//    K blocks are double-buffered in LDS via the async copy engine
//    (global_load_async_to_lds_b128 / s_wait_asynccnt): block j+1 copies
//    while block j runs WMMAs + online softmax; K-frags come from ds_load_b128.
// ---------------------------------------------------------------------------
__global__ void flash_attn_kernel(const _Float16* __restrict__ Qh,
                                  const _Float16* __restrict__ Kh,
                                  const _Float16* __restrict__ Vt,
                                  _Float16* __restrict__ ctx) {
#if HAVE_ASYNC_LDS
  __shared__ _Float16 klds[FA_NW * 2 * 32 * 64];   // double-buffered K tiles
#endif
  __shared__ _Float16 plds[FA_NW * 16 * 40];       // P staging (80B rows, 16B-aligned)
  const int wave  = threadIdx.x >> 5;
  const int lane  = threadIdx.x & 31;
  const int r16   = lane & 15;
  const int half_ = lane >> 4;
  const int bh = blockIdx.y;
  const int b  = bh >> 4, h = bh & 15;
  const int q0 = (blockIdx.x * FA_NW + wave) * 16;

  const _Float16* qrow  = Qh + (size_t)(b * S_LEN + q0 + r16) * D_MODEL + h * HEAD_DIM;
  const _Float16* kbase = Kh + (size_t)b * S_LEN * D_MODEL + h * HEAD_DIM;
  const _Float16* vbase = Vt + (size_t)bh * HEAD_DIM * S_LEN;
  _Float16* myp = &plds[wave * 16 * 40];
#if HAVE_ASYNC_LDS
  _Float16* kb0 = &klds[wave * 2 * 32 * 64];
  _Float16* kb1 = kb0 + 32 * 64;
#endif

  const half16 qa0 = load_frag_a(qrow, 0, half_);
  const half16 qa1 = load_frag_a(qrow, 32, half_);

  float mrow[8], lrow[8];
  float8 o0 = {}, o1 = {}, o2 = {}, o3 = {};
#pragma unroll
  for (int v = 0; v < 8; v++) { mrow[v] = -1e30f; lrow[v] = 0.0f; }

  const int nblk = ((q0 + 15) >> 5) + 1;
#if HAVE_ASYNC_LDS
  async_copy_k(kbase, 0, kb0, lane);
#endif

  for (int blk = 0; blk < nblk; blk++) {
    const int j0 = blk << 5;
    // ---- K fragments for this 32-col block
#if HAVE_ASYNC_LDS
    _Float16* cur = (blk & 1) ? kb1 : kb0;
    _Float16* nxt = (blk & 1) ? kb0 : kb1;
    wait_async0();                                    // block blk resident
    if (blk + 1 < nblk) async_copy_k(kbase, j0 + 32, nxt, lane);
    const _Float16* kr0 = cur + r16 * 64;             // K row j0+r16 (64 halves)
    const _Float16* kr1 = cur + (16 + r16) * 64;      // K row j0+16+r16
#else
    const _Float16* kr0 = kbase + (size_t)(j0 + r16) * D_MODEL;
    const _Float16* kr1 = kbase + (size_t)(j0 + 16 + r16) * D_MODEL;
#endif
    // ---- scores: two 16x16 tiles (K-dim = head dim 64)
    float8 s0 = {}, s1 = {};
    s0 = wmma_f16(qa0, load_frag_b(kr0, 0, half_), s0);
    s0 = wmma_f16(qa1, load_frag_b(kr0, 32, half_), s0);
    s1 = wmma_f16(qa0, load_frag_b(kr1, 0, half_), s1);
    s1 = wmma_f16(qa1, load_frag_b(kr1, 32, half_), s1);

    const int jc0 = j0 + r16;        // column index of tile0 for this lane
    const int jc1 = j0 + 16 + r16;   // column index of tile1
#pragma unroll
    for (int v = 0; v < 8; v++) {
      const int q = q0 + 8 * half_ + v;           // row handled by (half_, v)
      float a0 = (jc0 <= q) ? s0[v] * 0.125f : -1e30f;
      float a1 = (jc1 <= q) ? s1[v] * 0.125f : -1e30f;
      // row-max across the 16 lanes of this half-wave
      float t = fmaxf(a0, a1);
      t = fmaxf(t, __shfl_xor(t, 1));
      t = fmaxf(t, __shfl_xor(t, 2));
      t = fmaxf(t, __shfl_xor(t, 4));
      t = fmaxf(t, __shfl_xor(t, 8));
      const float mnew  = fmaxf(mrow[v], t);
      const float alpha = __expf(mrow[v] - mnew);
      const float p0 = __expf(a0 - mnew);
      const float p1 = __expf(a1 - mnew);
      float rs = p0 + p1;
      rs += __shfl_xor(rs, 1);
      rs += __shfl_xor(rs, 2);
      rs += __shfl_xor(rs, 4);
      rs += __shfl_xor(rs, 8);
      lrow[v] = lrow[v] * alpha + rs;
      mrow[v] = mnew;
      o0[v] *= alpha; o1[v] *= alpha; o2[v] *= alpha; o3[v] *= alpha;
      // stage P (C-layout -> row-major 16x32 in LDS)
      const int rr = 8 * half_ + v;
      myp[rr * 40 + r16]      = (_Float16)p0;
      myp[rr * 40 + 16 + r16] = (_Float16)p1;
    }
    // ---- P @ V over this 32-col block (K-dim = 32 keys), 4 d-tiles of 16
    const half16 pa = load_frag_a(myp + r16 * 40, 0, half_);
    o0 = wmma_f16(pa, load_frag_b(vbase + (size_t)(r16)      * S_LEN + j0, 0, half_), o0);
    o1 = wmma_f16(pa, load_frag_b(vbase + (size_t)(16 + r16) * S_LEN + j0, 0, half_), o1);
    o2 = wmma_f16(pa, load_frag_b(vbase + (size_t)(32 + r16) * S_LEN + j0, 0, half_), o2);
    o3 = wmma_f16(pa, load_frag_b(vbase + (size_t)(48 + r16) * S_LEN + j0, 0, half_), o3);
  }

  // ---- normalize and store context (f16, [b*S+s][h*64+d])
#pragma unroll
  for (int v = 0; v < 8; v++) {
    const float inv = 1.0f / lrow[v];
    const int r = 8 * half_ + v;
    size_t base = (size_t)(b * S_LEN + q0 + r) * D_MODEL + h * HEAD_DIM + r16;
    ctx[base]      = (_Float16)(o0[v] * inv);
    ctx[base + 16] = (_Float16)(o1[v] * inv);
    ctx[base + 32] = (_Float16)(o2[v] * inv);
    ctx[base + 48] = (_Float16)(o3[v] * inv);
  }
}

// ---------------------------------------------------------------------------
// Launch
// ---------------------------------------------------------------------------
extern "C" void kernel_launch(void* const* d_in, const int* in_sizes, int n_in,
                              void* d_out, int out_size, void* d_ws, size_t ws_size,
                              hipStream_t stream) {
  const float* x  = (const float*)d_in[0];
  const float* Wq = (const float*)d_in[1];
  const float* Wk = (const float*)d_in[2];
  const float* Wv = (const float*)d_in[3];
  const float* Wo = (const float*)d_in[4];
  // d_in[5] = attn_mask (causal; implemented analytically)
  float* out = (float*)d_out;

  char* ws = (char*)d_ws;
  size_t off = 0;
  auto carve = [&](size_t bytes) -> void* {
    void* p = ws + off;
    off += (bytes + 255) & ~(size_t)255;
    return p;
  };

  const size_t XD = (size_t)MTOT * D_MODEL;      // 4096*1024
  const size_t WD = (size_t)D_MODEL * D_MODEL;   // 1024*1024

  _Float16* xh  = (_Float16*)carve(XD * 2);
  _Float16* Wqh = (_Float16*)carve(WD * 2);
  _Float16* Wkh = (_Float16*)carve(WD * 2);
  _Float16* Wvh = (_Float16*)carve(WD * 2);
  _Float16* Woh = (_Float16*)carve(WD * 2);
  float*    Qf  = (float*)   carve(XD * 4);
  float*    Kf  = (float*)   carve(XD * 4);
  float*    Vf  = (float*)   carve(XD * 4);
  _Float16* Qh  = (_Float16*)carve(XD * 2);
  _Float16* Khh = (_Float16*)carve(XD * 2);
  _Float16* Vt  = (_Float16*)carve(XD * 2);
  _Float16* ctx = (_Float16*)carve(XD * 2);

  // 1) convert inputs to f16
  cvt_f32_f16<<<(int)(XD / 256), 256, 0, stream>>>(x,  xh,  (int)XD);
  cvt_f32_f16<<<(int)(WD / 256), 256, 0, stream>>>(Wq, Wqh, (int)WD);
  cvt_f32_f16<<<(int)(WD / 256), 256, 0, stream>>>(Wk, Wkh, (int)WD);
  cvt_f32_f16<<<(int)(WD / 256), 256, 0, stream>>>(Wv, Wvh, (int)WD);
  cvt_f32_f16<<<(int)(WD / 256), 256, 0, stream>>>(Wo, Woh, (int)WD);

  // 2) Q/K/V projections: [4096x1024] = xh @ W^T
  //    block tile 64(M) x 256(N); wave tile 32x64
  dim3 ggrid(D_MODEL / 256, MTOT / 64);
  gemm_wmma<<<ggrid, 256, 0, stream>>>(xh, Wqh, Qf, MTOT, D_MODEL, D_MODEL);
  gemm_wmma<<<ggrid, 256, 0, stream>>>(xh, Wkh, Kf, MTOT, D_MODEL, D_MODEL);
  gemm_wmma<<<ggrid, 256, 0, stream>>>(xh, Wvh, Vf, MTOT, D_MODEL, D_MODEL);

  // 3) RoPE on Q,K -> f16
  {
    int n = B_SZ * S_LEN * N_HEADS * 32;
    rope_kernel<<<n / 256, 256, 0, stream>>>(Qf, Kf, Qh, Khh);
  }
  // 4) transpose V -> Vt f16
  vtrans_kernel<<<(int)(XD / 256), 256, 0, stream>>>(Vf, Vt);

  // 5) flash attention
  {
    dim3 fgrid(S_LEN / 16 / FA_NW, B_SZ * N_HEADS);
    flash_attn_kernel<<<fgrid, 32 * FA_NW, 0, stream>>>(Qh, Khh, Vt, ctx);
  }

  // 6) output projection: out = ctx @ Wo^T (f32 result)
  gemm_wmma<<<ggrid, 256, 0, stream>>>(ctx, Woh, out, MTOT, D_MODEL, D_MODEL);
}
